// Attention_6201932775733
// MI455X (gfx1250) — matrix operand order repaired and checked by reading, not compiled, not run on hardware
//
#include <hip/hip_runtime.h>
#include <hip/hip_bf16.h>
#include <stdint.h>

// ---------------------------------------------------------------------------
// CDNA5 / gfx1250 windowed-attention block, bf16 WMMA (v_wmma_f32_16x16x32_bf16)
// Register-blocked GEMMs: 32(M) x 64(N) per wave, 8 WMMAs per K-step.
// ---------------------------------------------------------------------------

typedef __bf16 bf16;
typedef __attribute__((ext_vector_type(16))) __bf16 bf16x16;
typedef __attribute__((ext_vector_type(8)))  float  f32x8;

union FragU { bf16x16 v; uint4 q[2]; };
union PackU { uint4 q; bf16 h[8]; };

__device__ __forceinline__ f32x8 wmma_bf16(bf16x16 a, bf16x16 b, f32x8 c) {
  // 8 args: (neg_a, A, neg_b, B, c_mod, C, reuse_a, reuse_b)
  return __builtin_amdgcn_wmma_f32_16x16x32_bf16(false, a, false, b, (short)0, c,
                                                 false, false);
}

// A fragment: 16x32 bf16, source row-major [row][k], ld in elements.
// lane L holds row M = (L&15); lanes<16: K={kb+0..7, kb+16..23}; lanes>=16: +8.
__device__ __forceinline__ bf16x16 load_a_frag(const bf16* base, int row, int ld,
                                               int kbase, int lane) {
  FragU f;
  const bf16* p = base + (size_t)row * ld + kbase + ((lane & 16) ? 8 : 0);
  f.q[0] = *(const uint4*)(p);
  f.q[1] = *(const uint4*)(p + 16);
  return f.v;
}

// B fragment: 32x16 bf16, source stored as [col][k] (each column's K contiguous).
// lane L holds col N = (L&15); lanes<16: K={kb+0..7, kb+16..23}; lanes>=16: +8.
__device__ __forceinline__ bf16x16 load_b_frag(const bf16* base, int col, int ld,
                                               int kbase, int lane) {
  FragU f;
  const bf16* p = base + (size_t)col * ld + kbase + ((lane & 16) ? 8 : 0);
  f.q[0] = *(const uint4*)(p);
  f.q[1] = *(const uint4*)(p + 16);
  return f.v;
}

// ---------------------------------------------------------------------------
// Kernel 0: pack weights to bf16.  Wqkvb = [Wq(512) ; Wkv(1024)] x 256.
// ---------------------------------------------------------------------------
__global__ void pack_w_kernel(const float* __restrict__ Wq,
                              const float* __restrict__ Wkv,
                              const float* __restrict__ Wout,
                              bf16* __restrict__ Wqkvb,
                              bf16* __restrict__ Woutb) {
  int i = blockIdx.x * 256 + threadIdx.x;
  if (i < 1536 * 256)
    Wqkvb[i] = (bf16)((i < 512 * 256) ? Wq[i] : Wkv[i - 512 * 256]);
  if (i < 256 * 512)
    Woutb[i] = (bf16)Wout[i];
}

// ---------------------------------------------------------------------------
// Kernel 1: transpose (c,p)->(p,c), add time_emb to x, convert to bf16.
// xt/st: [b][16384 pixels][256 ch]
// ---------------------------------------------------------------------------
__global__ void prep_x_kernel(const float* __restrict__ x,
                              const float* __restrict__ skip,
                              const float* __restrict__ te,
                              bf16* __restrict__ xt, bf16* __restrict__ st) {
  __shared__ float tile[32][33];
  int b  = blockIdx.z;
  int c0 = blockIdx.y * 32;
  int p0 = blockIdx.x * 32;
  int tx = threadIdx.x & 31;
  int ty = threadIdx.x >> 5;  // 0..7
#pragma unroll
  for (int pass = 0; pass < 2; ++pass) {
    const float* src = pass ? skip : x;
#pragma unroll
    for (int r = ty; r < 32; r += 8) {
      float v = src[((size_t)b * 256 + c0 + r) * 16384 + p0 + tx];
      if (!pass) v += te[b * 256 + c0 + r];
      tile[r][tx] = v;
    }
    __syncthreads();
    bf16* dst = pass ? st : xt;
#pragma unroll
    for (int r = ty; r < 32; r += 8)
      dst[((size_t)b * 16384 + p0 + r) * 256 + c0 + tx] = (bf16)tile[tx][r];
    __syncthreads();
  }
}

// ---------------------------------------------------------------------------
// Kernel 2: fused QKV projection GEMM + RoPE + window scatter.
//   D[o][p] = sum_c W[o][c] * X[p][c]     (M=Mtot, N=16384 px, K=256, per batch)
//   src==0: o<512 -> q, <1024 -> k, else v  (x input, W = [Wq;Wkv])
//   src==1: o<512 -> k, else v              (skip input, W = Wkv)
// Wave computes 32(M) x 64(N).  Outputs:
//   qbuf[win][t][64]   kbuf[win][src*256+t][64]   vT[win][64][512]
// ---------------------------------------------------------------------------
__device__ __forceinline__ void proj_epilogue(
    f32x8 acc, int Mbase, int Nbase, int batch, int src, int lane,
    const float* __restrict__ sinb, const float* __restrict__ cosb,
    bf16* __restrict__ qbuf, bf16* __restrict__ kbuf, bf16* __restrict__ vT) {
  int hi  = (lane & 16) ? 8 : 0;
  int p   = Nbase + (lane & 15);
  int Xr  = p >> 7, Yc = p & 127;
  int nxi = Xr >> 4, wx = Xr & 15, nyi = Yc >> 4, wy = Yc & 15;
  int t   = wx * 16 + wy;
  int oi  = Mbase & 511;
  int head = oi >> 6;
  int db  = (oi & 63) + hi;                     // head-dim base for register 0
  int win = ((batch * 8 + head) * 8 + nxi) * 8 + nyi;
  int regn = (src == 0) ? (Mbase >> 9) : 1 + (Mbase >> 9);  // 0=q 1=k 2=v

  float vals[8];
#pragma unroll
  for (int j = 0; j < 8; ++j) vals[j] = acc[j];

  if (regn <= 1 && (Mbase & 63) < 32) {         // RoPE region (d < 32) for q/k
    const float* cp = cosb + ((size_t)p * 32 + db);
    const float* sp = sinb + ((size_t)p * 32 + db);
    float4 c0 = *(const float4*)(cp), c1 = *(const float4*)(cp + 4);
    float4 s0 = *(const float4*)(sp), s1 = *(const float4*)(sp + 4);
    float cc[8] = {c0.x, c0.y, c0.z, c0.w, c1.x, c1.y, c1.z, c1.w};
    float ss[8] = {s0.x, s0.y, s0.z, s0.w, s1.x, s1.y, s1.z, s1.w};
#pragma unroll
    for (int j = 0; j < 8; j += 2) {
      float e = vals[j], o = vals[j + 1];
      vals[j]     = e * cc[j]     - o * ss[j];
      vals[j + 1] = o * cc[j + 1] + e * ss[j + 1];
    }
  }

  if (regn == 0) {
    PackU pk;
#pragma unroll
    for (int j = 0; j < 8; ++j) pk.h[j] = (bf16)vals[j];
    *(uint4*)(qbuf + ((size_t)win * 256 + t) * 64 + db) = pk.q;
  } else if (regn == 1) {
    PackU pk;
#pragma unroll
    for (int j = 0; j < 8; ++j) pk.h[j] = (bf16)vals[j];
    *(uint4*)(kbuf + ((size_t)win * 512 + src * 256 + t) * 64 + db) = pk.q;
  } else {                                      // v, transposed: vT[win][d][tok]
    int tok = src * 256 + t;
#pragma unroll
    for (int j = 0; j < 8; ++j)
      vT[((size_t)win * 64 + db + j) * 512 + tok] = (bf16)vals[j];
  }
}

__global__ void proj_kernel(const bf16* __restrict__ Xsrc,
                            const bf16* __restrict__ W,
                            const float* __restrict__ sinb,
                            const float* __restrict__ cosb,
                            bf16* __restrict__ qbuf, bf16* __restrict__ kbuf,
                            bf16* __restrict__ vT, int Mtot, int src) {
  int lane = threadIdx.x & 31;
  int wave = threadIdx.x >> 5;
  int gid = blockIdx.x * 8 + wave;
  int mblk = Mtot >> 5;                 // 32-row M blocks
  int tpb  = mblk * 256;                // 64-col N blocks: 16384/64 = 256
  int batch = gid / tpb;
  int r     = gid % tpb;
  int Mbase = (r / 256) * 32;
  int Nbase = (r % 256) * 64;

  const bf16* Xb = Xsrc + (size_t)batch * 16384 * 256;
  f32x8 acc[2][4];
#pragma unroll
  for (int mi = 0; mi < 2; ++mi)
#pragma unroll
    for (int ni = 0; ni < 4; ++ni) acc[mi][ni] = (f32x8){0,0,0,0,0,0,0,0};

#pragma unroll
  for (int kb = 0; kb < 256; kb += 32) {
    bf16x16 a[2], b[4];
#pragma unroll
    for (int mi = 0; mi < 2; ++mi)
      a[mi] = load_a_frag(W, Mbase + mi * 16 + (lane & 15), 256, kb, lane);
#pragma unroll
    for (int ni = 0; ni < 4; ++ni)
      b[ni] = load_b_frag(Xb, Nbase + ni * 16 + (lane & 15), 256, kb, lane);
#pragma unroll
    for (int mi = 0; mi < 2; ++mi)
#pragma unroll
      for (int ni = 0; ni < 4; ++ni)
        acc[mi][ni] = wmma_bf16(a[mi], b[ni], acc[mi][ni]);
  }

#pragma unroll
  for (int mi = 0; mi < 2; ++mi)
#pragma unroll
    for (int ni = 0; ni < 4; ++ni)
      proj_epilogue(acc[mi][ni], Mbase + mi * 16, Nbase + ni * 16, batch, src,
                    lane, sinb, cosb, qbuf, kbuf, vT);
}

// ---------------------------------------------------------------------------
// Kernel 3: flash attention per window. 1 block (8 waves) per window.
// wave handles 32 q-rows; loops over 8 blocks of 64 kv-columns.
// obuf: [b][pixel][512 inner-ch] bf16 for the output projection.
// ---------------------------------------------------------------------------
__global__ void attn_kernel(const bf16* __restrict__ qbuf,
                            const bf16* __restrict__ kbuf,
                            const bf16* __restrict__ vT,
                            bf16* __restrict__ obuf) {
  __shared__ __align__(16) bf16 pshare[8 * 32 * 64];
  int win  = blockIdx.x;
  int lane = threadIdx.x & 31;
  int wave = threadIdx.x >> 5;
  bf16* pwave = pshare + wave * (32 * 64);

  const bf16* qb = qbuf + (size_t)win * 256 * 64;
  const bf16* kb = kbuf + (size_t)win * 512 * 64;
  const bf16* vb = vT   + (size_t)win * 64 * 512;
  int qr0 = wave * 32;

  bf16x16 aQ[2][2];
#pragma unroll
  for (int mt = 0; mt < 2; ++mt)
#pragma unroll
    for (int kt = 0; kt < 2; ++kt)
      aQ[mt][kt] = load_a_frag(qb, qr0 + mt * 16 + (lane & 15), 64, kt * 32, lane);

  f32x8 O[2][4];
  float mstat[2][8], lstat[2][8];
#pragma unroll
  for (int mt = 0; mt < 2; ++mt) {
#pragma unroll
    for (int dt = 0; dt < 4; ++dt) O[mt][dt] = (f32x8){0,0,0,0,0,0,0,0};
#pragma unroll
    for (int j = 0; j < 8; ++j) { mstat[mt][j] = -1e30f; lstat[mt][j] = 0.f; }
  }
  const float scale = 0.125f;  // dh^-0.5, dh=64
  int hi = (lane & 16) ? 8 : 0;

  for (int cb = 0; cb < 512; cb += 64) {
    // ---- S = scale * Q K^T for 32x64 block ----
    f32x8 S[2][4];
#pragma unroll
    for (int mt = 0; mt < 2; ++mt)
#pragma unroll
      for (int nt = 0; nt < 4; ++nt) S[mt][nt] = (f32x8){0,0,0,0,0,0,0,0};
#pragma unroll
    for (int nt = 0; nt < 4; ++nt)
#pragma unroll
      for (int kt = 0; kt < 2; ++kt) {
        bf16x16 bK = load_b_frag(kb, cb + nt * 16 + (lane & 15), 64, kt * 32, lane);
#pragma unroll
        for (int mt = 0; mt < 2; ++mt)
          S[mt][nt] = wmma_bf16(aQ[mt][kt], bK, S[mt][nt]);
      }
#pragma unroll
    for (int mt = 0; mt < 2; ++mt)
#pragma unroll
      for (int nt = 0; nt < 4; ++nt)
#pragma unroll
        for (int j = 0; j < 8; ++j) S[mt][nt][j] *= scale;

    // ---- online softmax: row stats across the 16-lane N groups ----
#pragma unroll
    for (int mt = 0; mt < 2; ++mt) {
#pragma unroll
      for (int j = 0; j < 8; ++j) {
        float rm = fmaxf(fmaxf(S[mt][0][j], S[mt][1][j]),
                         fmaxf(S[mt][2][j], S[mt][3][j]));
#pragma unroll
        for (int m = 1; m < 16; m <<= 1) rm = fmaxf(rm, __shfl_xor(rm, m, 32));
        float mnew = fmaxf(mstat[mt][j], rm);
        float corr = __expf(mstat[mt][j] - mnew);
        mstat[mt][j] = mnew;
        float rs = 0.f;
#pragma unroll
        for (int nt = 0; nt < 4; ++nt) {
          float pv = __expf(S[mt][nt][j] - mnew);
          S[mt][nt][j] = pv;
          rs += pv;
        }
#pragma unroll
        for (int m = 1; m < 16; m <<= 1) rs += __shfl_xor(rs, m, 32);
        lstat[mt][j] = lstat[mt][j] * corr + rs;
#pragma unroll
        for (int dt = 0; dt < 4; ++dt) O[mt][dt][j] *= corr;
      }
      // P (D layout) -> per-wave LDS tile (row-major 32x64) for A-fragment reload
#pragma unroll
      for (int nt = 0; nt < 4; ++nt)
#pragma unroll
        for (int j = 0; j < 8; ++j)
          pwave[(mt * 16 + j + hi) * 64 + nt * 16 + (lane & 15)] =
              (bf16)S[mt][nt][j];
    }
    asm volatile("s_wait_dscnt 0" ::: "memory");

    // ---- O += P V ----
    bf16x16 aP[2][2];
#pragma unroll
    for (int mt = 0; mt < 2; ++mt)
#pragma unroll
      for (int kt = 0; kt < 2; ++kt)
        aP[mt][kt] = load_a_frag(pwave, mt * 16 + (lane & 15), 64, kt * 32, lane);
#pragma unroll
    for (int dt = 0; dt < 4; ++dt)
#pragma unroll
      for (int kt = 0; kt < 2; ++kt) {
        bf16x16 bV = load_b_frag(vb, dt * 16 + (lane & 15), 512, cb + kt * 32, lane);
#pragma unroll
        for (int mt = 0; mt < 2; ++mt)
          O[mt][dt] = wmma_bf16(aP[mt][kt], bV, O[mt][dt]);
      }
  }

  // ---- finalize and scatter to [b][pixel][inner-ch] ----
  int b = win >> 9, h = (win >> 6) & 7, nxi = (win >> 3) & 7, nyi = win & 7;
#pragma unroll
  for (int mt = 0; mt < 2; ++mt)
#pragma unroll
    for (int j = 0; j < 8; ++j) {
      float inv = 1.0f / lstat[mt][j];
      int token = qr0 + mt * 16 + j + hi;
      int wx = token >> 4, wy = token & 15;
      int p = (nxi * 16 + wx) * 128 + (nyi * 16 + wy);
      bf16* dst = obuf + ((size_t)b * 16384 + p) * 512 + h * 64;
#pragma unroll
      for (int dt = 0; dt < 4; ++dt)
        dst[dt * 16 + (lane & 15)] = (bf16)(O[mt][dt][j] * inv);
    }
}

// ---------------------------------------------------------------------------
// Kernel 4: output projection.  out[b][o][p] = sum_c Wout[o][c]*obuf[b][p][c] + bout
//   M=256, K=512, N=16384 per batch.  Wave computes 32(M) x 64(N).
// ---------------------------------------------------------------------------
__global__ void outproj_kernel(const bf16* __restrict__ obuf,
                               const bf16* __restrict__ Woutb,
                               const float* __restrict__ bout,
                               float* __restrict__ out) {
  int lane = threadIdx.x & 31;
  int wave = threadIdx.x >> 5;
  int gid = blockIdx.x * 8 + wave;
  int tpb = 8 * 256;                    // (256/32) M blocks * (16384/64) N blocks
  int batch = gid / tpb;
  int r     = gid % tpb;
  int Mbase = (r / 256) * 32;
  int Nbase = (r % 256) * 64;

  const bf16* Ob = obuf + (size_t)batch * 16384 * 512;
  f32x8 acc[2][4];
#pragma unroll
  for (int mi = 0; mi < 2; ++mi)
#pragma unroll
    for (int ni = 0; ni < 4; ++ni) acc[mi][ni] = (f32x8){0,0,0,0,0,0,0,0};

#pragma unroll
  for (int kb = 0; kb < 512; kb += 32) {
    bf16x16 a[2], b[4];
#pragma unroll
    for (int mi = 0; mi < 2; ++mi)
      a[mi] = load_a_frag(Woutb, Mbase + mi * 16 + (lane & 15), 512, kb, lane);
#pragma unroll
    for (int ni = 0; ni < 4; ++ni)
      b[ni] = load_b_frag(Ob, Nbase + ni * 16 + (lane & 15), 512, kb, lane);
#pragma unroll
    for (int mi = 0; mi < 2; ++mi)
#pragma unroll
      for (int ni = 0; ni < 4; ++ni)
        acc[mi][ni] = wmma_bf16(a[mi], b[ni], acc[mi][ni]);
  }

  int hi = (lane & 16) ? 8 : 0;
#pragma unroll
  for (int mi = 0; mi < 2; ++mi)
#pragma unroll
    for (int ni = 0; ni < 4; ++ni) {
      int p = Nbase + ni * 16 + (lane & 15);
#pragma unroll
      for (int j = 0; j < 8; ++j) {
        int o = Mbase + mi * 16 + hi + j;
        out[((size_t)batch * 256 + o) * 16384 + p] = acc[mi][ni][j] + bout[o];
      }
    }
}

// ---------------------------------------------------------------------------
// Launch
// ---------------------------------------------------------------------------
extern "C" void kernel_launch(void* const* d_in, const int* in_sizes, int n_in,
                              void* d_out, int out_size, void* d_ws, size_t ws_size,
                              hipStream_t stream) {
  const float* x    = (const float*)d_in[0];
  const float* skip = (const float*)d_in[1];
  const float* te   = (const float*)d_in[2];
  const float* sinb = (const float*)d_in[3];
  const float* cosb = (const float*)d_in[4];
  const float* Wq   = (const float*)d_in[5];
  const float* Wkv  = (const float*)d_in[6];
  const float* Wout = (const float*)d_in[7];
  const float* bout = (const float*)d_in[8];
  float* out = (float*)d_out;

  char* ws = (char*)d_ws;
  size_t off = 0;
  auto take = [&](size_t bytes) {
    void* p = ws + off;
    off = (off + bytes + 255) & ~(size_t)255;
    return p;
  };
  bf16* Wqkvb = (bf16*)take((size_t)1536 * 256 * 2);
  bf16* Woutb = (bf16*)take((size_t)256 * 512 * 2);
  bf16* xt    = (bf16*)take((size_t)2 * 16384 * 256 * 2);
  bf16* st    = (bf16*)take((size_t)2 * 16384 * 256 * 2);
  bf16* qb    = (bf16*)take((size_t)1024 * 256 * 64 * 2);
  bf16* kb    = (bf16*)take((size_t)1024 * 512 * 64 * 2);
  bf16* vt    = (bf16*)take((size_t)1024 * 64 * 512 * 2);
  bf16* ob    = (bf16*)take((size_t)2 * 16384 * 512 * 2);

  pack_w_kernel<<<1536, 256, 0, stream>>>(Wq, Wkv, Wout, Wqkvb, Woutb);
  prep_x_kernel<<<dim3(512, 8, 2), 256, 0, stream>>>(x, skip, te, xt, st);
  // x pass: M=1536 -> 48 Mblk * 256 Nblk * 2 batches = 24576 waves / 8
  proj_kernel<<<3072, 256, 0, stream>>>(xt, Wqkvb, sinb, cosb, qb, kb, vt, 1536, 0);
  // skip pass: M=1024 -> 32 Mblk * 256 Nblk * 2 batches = 16384 waves / 8
  proj_kernel<<<2048, 256, 0, stream>>>(st, Wqkvb + 512 * 256, sinb, cosb,
                                        qb, kb, vt, 1024, 1);
  attn_kernel<<<1024, 256, 0, stream>>>(qb, kb, vt, ob);
  // outproj: 8 Mblk * 256 Nblk * 2 batches = 4096 waves / 8
  outproj_kernel<<<512, 256, 0, stream>>>(ob, Woutb, bout, out);
}
